// MoE_87565793231316
// MI455X (gfx1250) — matrix operand order repaired and checked by reading, not compiled, not run on hardware
//
#include <hip/hip_runtime.h>

#define D_MODEL 1024
#define N_EXP   8
#define HID     2752
#define NTOK    4096
#define ALPHA   0.05f

#define BM 32
#define BN 64
#define BK 32
#define LDA 40   // staged row stride in bf16 units (>=32, multiple of 8 -> 16B aligned rows)

typedef float  v8f   __attribute__((ext_vector_type(8)));
typedef __bf16 v4bf  __attribute__((ext_vector_type(4)));
typedef __bf16 v8bf  __attribute__((ext_vector_type(8)));
typedef __bf16 v16bf __attribute__((ext_vector_type(16)));

#define WMMA_BF16(A, B, C) \
    __builtin_amdgcn_wmma_f32_16x16x32_bf16(false, (A), false, (B), (short)0, (C), false, false)

__device__ __forceinline__ void split_bf16(float v, __bf16* hi, __bf16* lo) {
    const __bf16 h = (__bf16)v;
    *hi = h;
    *lo = (__bf16)(v - (float)h);
}

// float4 -> hi/lo v4bf (for contiguous A staging, 8B LDS stores)
__device__ __forceinline__ void split4(const float4 f, v4bf* hi, v4bf* lo) {
    __bf16 h0, h1, h2, h3, l0, l1, l2, l3;
    split_bf16(f.x, &h0, &l0);
    split_bf16(f.y, &h1, &l1);
    split_bf16(f.z, &h2, &l2);
    split_bf16(f.w, &h3, &l3);
    v4bf h = {h0, h1, h2, h3};
    v4bf l = {l0, l1, l2, l3};
    *hi = h;
    *lo = l;
}

// float4 (4 consecutive N of one K-row) -> transposed hi/lo stores at column kk
__device__ __forceinline__ void store_tr4(const float4 f, __bf16* Hb, __bf16* Lb, int kk) {
    const float vals[4] = {f.x, f.y, f.z, f.w};
#pragma unroll
    for (int i = 0; i < 4; ++i) {
        __bf16 h, l;
        split_bf16(vals[i], &h, &l);
        Hb[i * LDA + kk] = h;
        Lb[i * LDA + kk] = l;
    }
}

__device__ __forceinline__ v16bf cat8(v8bf a, v8bf b) {
    return __builtin_shufflevector(a, b, 0, 1, 2, 3, 4, 5, 6, 7, 8, 9, 10, 11, 12, 13, 14, 15);
}

// A fragment (16x32 bf16): lane(half,m) holds K in [8h,8h+8) and [16+8h,16+8h+8)
__device__ __forceinline__ v16bf frag_a(const __bf16* row, int half) {
    const v8bf s0 = *(const v8bf*)(row + 8 * half);
    const v8bf s1 = *(const v8bf*)(row + 16 + 8 * half);
    return cat8(s0, s1);
}
// B fragment (32x16 bf16), staged column-major: lane(half,n) holds K in [8h,8h+8) and [16+8h,16+8h+8)
__device__ __forceinline__ v16bf frag_b(const __bf16* row, int half) {
    const v8bf s0 = *(const v8bf*)(row + 8 * half);
    const v8bf s1 = *(const v8bf*)(row + 16 + 8 * half);
    return cat8(s0, s1);
}

// ---------------- init ----------------
__global__ void moe_init_kernel(int* cnt) {
    if (threadIdx.x < N_EXP) cnt[threadIdx.x] = 0;
}

// ---------------- router: one wave32 per token ----------------
__global__ __launch_bounds__(256)
void moe_router_kernel(const float* __restrict__ x,
                       const float* __restrict__ Wg,
                       const float* __restrict__ bg,
                       int* __restrict__ cnt,
                       float* __restrict__ probs,
                       float* __restrict__ top1_p,
                       int* __restrict__ lists) {
    const int wave = threadIdx.x >> 5;
    const int lane = threadIdx.x & 31;
    const int tok  = blockIdx.x * 8 + wave;
    const float* xr = x + (size_t)tok * D_MODEL;

    float acc[N_EXP];
#pragma unroll
    for (int e = 0; e < N_EXP; ++e) acc[e] = 0.f;

    for (int k = lane; k < D_MODEL; k += 32) {
        const float xv = xr[k];
        const float* wr = Wg + (size_t)k * N_EXP;
#pragma unroll
        for (int e = 0; e < N_EXP; ++e) acc[e] += xv * wr[e];
    }
#pragma unroll
    for (int e = 0; e < N_EXP; ++e) {
#pragma unroll
        for (int off = 16; off > 0; off >>= 1)
            acc[e] += __shfl_down(acc[e], off, 32);
    }

    if (lane == 0) {
        float lg[N_EXP], p[N_EXP];
        float m = -1e30f;
#pragma unroll
        for (int e = 0; e < N_EXP; ++e) { lg[e] = acc[e] + bg[e]; m = fmaxf(m, lg[e]); }
        float s = 0.f;
#pragma unroll
        for (int e = 0; e < N_EXP; ++e) { p[e] = __expf(lg[e] - m); s += p[e]; }
        const float inv = 1.f / s;
        int best = 0; float bp = p[0] * inv;
#pragma unroll
        for (int e = 0; e < N_EXP; ++e) {
            const float pe = p[e] * inv;
            probs[(size_t)tok * N_EXP + e] = pe;
            if (pe > bp) { bp = pe; best = e; }
        }
        top1_p[tok] = bp;
        const int slot = atomicAdd(&cnt[best], 1);
        lists[(size_t)best * NTOK + slot] = tok;
    }
}

// ---------------- aux loss (deterministic reduction) ----------------
__global__ __launch_bounds__(256)
void moe_aux_kernel(const int* __restrict__ cnt,
                    const float* __restrict__ probs,
                    float* __restrict__ out_aux) {
    __shared__ float ce_sh[N_EXP];
    const int w = threadIdx.x >> 5;
    const int lane = threadIdx.x & 31;
    float s = 0.f;
    for (int i = lane; i < NTOK; i += 32) s += probs[(size_t)i * N_EXP + w];
#pragma unroll
    for (int off = 16; off > 0; off >>= 1) s += __shfl_down(s, off, 32);
    if (lane == 0) ce_sh[w] = s;
    __syncthreads();
    if (threadIdx.x == 0) {
        float a = 0.f;
        const float invN = 1.f / (float)NTOK;
#pragma unroll
        for (int e = 0; e < N_EXP; ++e)
            a += ((float)cnt[e] * invN) * (ce_sh[e] * invN);
        *out_aux = ALPHA * (float)N_EXP * a;
    }
}

// ---------------- pass A: h = silu(x @ Wu[e]) * (x @ Wv[e]) ----------------
__global__ __launch_bounds__(128)
void moe_up_kernel(const float* __restrict__ x,
                   const float* __restrict__ Wu,
                   const float* __restrict__ Wv,
                   const int* __restrict__ cnt,
                   const int* __restrict__ lists,
                   float* __restrict__ h_ws) {
    const int e  = blockIdx.z;
    const int me = cnt[e];
    const int m0 = blockIdx.y * BM;
    if (m0 >= me) return;                        // uniform exit
    const int n0 = blockIdx.x * BN;

    __shared__ alignas(16) __bf16 AsH[BM][LDA];
    __shared__ alignas(16) __bf16 AsL[BM][LDA];
    __shared__ alignas(16) __bf16 BuH[BN][LDA];
    __shared__ alignas(16) __bf16 BuL[BN][LDA];
    __shared__ alignas(16) __bf16 BvH[BN][LDA];
    __shared__ alignas(16) __bf16 BvL[BN][LDA];
    __shared__ int tks[BM];

    const int tid = threadIdx.x;
    if (tid < BM) {
        const int slot = m0 + tid;
        tks[tid] = (slot < me) ? lists[(size_t)e * NTOK + slot]
                               : lists[(size_t)e * NTOK];     // safe pad row
    }
    __syncthreads();

    const int lane = tid & 31;
    const int w    = tid >> 5;
    const int half = lane >> 4;
    const int nr   = lane & 15;
    const int nc   = w * 16 + nr;

    const float* WuE = Wu + (size_t)e * D_MODEL * HID;
    const float* WvE = Wv + (size_t)e * D_MODEL * HID;

    // Loop-invariant staging slots: A = 2 float4/thread, B = 4 float4/thread/matrix
    const int am  = tid >> 3;                 // A row 0..15 (and +16)
    const int ak  = (tid & 7) << 2;           // A col-within-chunk (float4)
    const float* aptr0 = x + (size_t)tks[am] * D_MODEL + ak;
    const float* aptr1 = x + (size_t)tks[am + 16] * D_MODEL + ak;

    const int bk  = tid >> 4;                 // B base row 0..7 (+8j)
    const int bn  = (tid & 15) << 2;          // B col group
    const float* buptr = WuE + (size_t)bk * HID + n0 + bn;
    const float* bvptr = WvE + (size_t)bk * HID + n0 + bn;

    v8f u0 = {}, u1 = {}, q0 = {}, q1 = {};

    for (int kc = 0; kc < D_MODEL; kc += BK) {
        // stage A (gathered token rows): two independent b128 loads, b64 LDS stores
        {
            const float4 fa0 = *(const float4*)(aptr0 + kc);
            const float4 fa1 = *(const float4*)(aptr1 + kc);
            v4bf h, l;
            split4(fa0, &h, &l);
            *(v4bf*)&AsH[am][ak] = h;       *(v4bf*)&AsL[am][ak] = l;
            split4(fa1, &h, &l);
            *(v4bf*)&AsH[am + 16][ak] = h;  *(v4bf*)&AsL[am + 16][ak] = l;
        }
        // stage Wu/Wv: 8 independent b128 loads, transposed hi/lo bf16 stores
#pragma unroll
        for (int j = 0; j < 4; ++j) {
            const size_t go = (size_t)(kc + 8 * j) * HID;
            const float4 fu = *(const float4*)(buptr + go);
            const float4 fv = *(const float4*)(bvptr + go);
            const int kk = bk + 8 * j;
            store_tr4(fu, &BuH[bn][0], &BuL[bn][0], kk);
            store_tr4(fv, &BvH[bn][0], &BvL[bn][0], kk);
        }
        __syncthreads();

        // prefetch next K-chunk (global_prefetch), one request per 64B line
        if (kc + BK < D_MODEL) {
            const size_t pg = (size_t)(kc + BK + (tid >> 2)) * HID + n0 + ((tid & 3) << 4);
            __builtin_prefetch(WuE + pg, 0, 3);
            __builtin_prefetch(WvE + pg, 0, 3);
            if ((tid & 3) == 0) {
                __builtin_prefetch(aptr0 + kc + BK, 0, 3);
                __builtin_prefetch(aptr1 + kc + BK, 0, 3);
            }
        }

        const v16bf a0h = frag_a(&AsH[nr][0], half);
        const v16bf a0l = frag_a(&AsL[nr][0], half);
        const v16bf a1h = frag_a(&AsH[16 + nr][0], half);
        const v16bf a1l = frag_a(&AsL[16 + nr][0], half);
        {
            const v16bf bh = frag_b(&BuH[nc][0], half);
            const v16bf bl = frag_b(&BuL[nc][0], half);
            u0 = WMMA_BF16(a0h, bh, u0);  u1 = WMMA_BF16(a1h, bh, u1);
            u0 = WMMA_BF16(a0l, bh, u0);  u1 = WMMA_BF16(a1l, bh, u1);
            u0 = WMMA_BF16(a0h, bl, u0);  u1 = WMMA_BF16(a1h, bl, u1);
        }
        {
            const v16bf bh = frag_b(&BvH[nc][0], half);
            const v16bf bl = frag_b(&BvL[nc][0], half);
            q0 = WMMA_BF16(a0h, bh, q0);  q1 = WMMA_BF16(a1h, bh, q1);
            q0 = WMMA_BF16(a0l, bh, q0);  q1 = WMMA_BF16(a1l, bh, q1);
            q0 = WMMA_BF16(a0h, bl, q0);  q1 = WMMA_BF16(a1h, bl, q1);
        }
        __syncthreads();
    }

    const int col = n0 + nc;
#pragma unroll
    for (int r = 0; r < 8; ++r) {
        const int row0 = r + 8 * half;
        const int row1 = 16 + row0;
        float uu = u0[r], vv = q0[r];
        float hv = (uu / (1.f + __expf(-uu))) * vv;
        if (m0 + row0 < me) h_ws[(size_t)tks[row0] * HID + col] = hv;
        uu = u1[r]; vv = q1[r];
        hv = (uu / (1.f + __expf(-uu))) * vv;
        if (m0 + row1 < me) h_ws[(size_t)tks[row1] * HID + col] = hv;
    }
}

// ---------------- pass B: y = top1_p * (h @ Wd[e]) ----------------
__global__ __launch_bounds__(128)
void moe_down_kernel(const float* __restrict__ h_ws,
                     const float* __restrict__ Wd,
                     const int* __restrict__ cnt,
                     const int* __restrict__ lists,
                     const float* __restrict__ top1_p,
                     float* __restrict__ y) {
    const int e  = blockIdx.z;
    const int me = cnt[e];
    const int m0 = blockIdx.y * BM;
    if (m0 >= me) return;
    const int n0 = blockIdx.x * BN;

    __shared__ alignas(16) __bf16 AsH[BM][LDA];
    __shared__ alignas(16) __bf16 AsL[BM][LDA];
    __shared__ alignas(16) __bf16 BdH[BN][LDA];
    __shared__ alignas(16) __bf16 BdL[BN][LDA];
    __shared__ int tks[BM];

    const int tid = threadIdx.x;
    if (tid < BM) {
        const int slot = m0 + tid;
        tks[tid] = (slot < me) ? lists[(size_t)e * NTOK + slot]
                               : lists[(size_t)e * NTOK];
    }
    __syncthreads();

    const int lane = tid & 31;
    const int w    = tid >> 5;
    const int half = lane >> 4;
    const int nr   = lane & 15;
    const int nc   = w * 16 + nr;

    const float* WdE = Wd + (size_t)e * HID * D_MODEL;

    const int am  = tid >> 3;
    const int ak  = (tid & 7) << 2;
    const float* aptr0 = h_ws + (size_t)tks[am] * HID + ak;
    const float* aptr1 = h_ws + (size_t)tks[am + 16] * HID + ak;

    const int bk  = tid >> 4;
    const int bn  = (tid & 15) << 2;
    const float* bdptr = WdE + (size_t)bk * D_MODEL + n0 + bn;

    v8f c0 = {}, c1 = {};

    for (int kc = 0; kc < HID; kc += BK) {       // 86 chunks
        {
            const float4 fa0 = *(const float4*)(aptr0 + kc);
            const float4 fa1 = *(const float4*)(aptr1 + kc);
            v4bf h, l;
            split4(fa0, &h, &l);
            *(v4bf*)&AsH[am][ak] = h;       *(v4bf*)&AsL[am][ak] = l;
            split4(fa1, &h, &l);
            *(v4bf*)&AsH[am + 16][ak] = h;  *(v4bf*)&AsL[am + 16][ak] = l;
        }
#pragma unroll
        for (int j = 0; j < 4; ++j) {
            const float4 fd = *(const float4*)(bdptr + (size_t)(kc + 8 * j) * D_MODEL);
            store_tr4(fd, &BdH[bn][0], &BdL[bn][0], bk + 8 * j);
        }
        __syncthreads();

        if (kc + BK < HID) {
            const size_t pg = (size_t)(kc + BK + (tid >> 2)) * D_MODEL + n0 + ((tid & 3) << 4);
            __builtin_prefetch(WdE + pg, 0, 3);
            if ((tid & 3) == 0) {
                __builtin_prefetch(aptr0 + kc + BK, 0, 3);
                __builtin_prefetch(aptr1 + kc + BK, 0, 3);
            }
        }

        const v16bf a0h = frag_a(&AsH[nr][0], half);
        const v16bf a0l = frag_a(&AsL[nr][0], half);
        const v16bf a1h = frag_a(&AsH[16 + nr][0], half);
        const v16bf a1l = frag_a(&AsL[16 + nr][0], half);
        const v16bf bh  = frag_b(&BdH[nc][0], half);
        const v16bf bl  = frag_b(&BdL[nc][0], half);
        c0 = WMMA_BF16(a0h, bh, c0);  c1 = WMMA_BF16(a1h, bh, c1);
        c0 = WMMA_BF16(a0l, bh, c0);  c1 = WMMA_BF16(a1l, bh, c1);
        c0 = WMMA_BF16(a0h, bl, c0);  c1 = WMMA_BF16(a1h, bl, c1);
        __syncthreads();
    }

    const int col = n0 + nc;
#pragma unroll
    for (int r = 0; r < 8; ++r) {
        const int row0 = r + 8 * half;
        const int row1 = 16 + row0;
        if (m0 + row0 < me) {
            const int t = tks[row0];
            y[(size_t)t * D_MODEL + col] = top1_p[t] * c0[r];
        }
        if (m0 + row1 < me) {
            const int t = tks[row1];
            y[(size_t)t * D_MODEL + col] = top1_p[t] * c1[r];
        }
    }
}

// ---------------- host ----------------
extern "C" void kernel_launch(void* const* d_in, const int* in_sizes, int n_in,
                              void* d_out, int out_size, void* d_ws, size_t ws_size,
                              hipStream_t stream) {
    const float* x  = (const float*)d_in[0];
    const float* Wg = (const float*)d_in[1];
    const float* bg = (const float*)d_in[2];
    const float* Wu = (const float*)d_in[3];
    const float* Wv = (const float*)d_in[4];
    const float* Wd = (const float*)d_in[5];
    float* out = (float*)d_out;

    char* ws = (char*)d_ws;
    size_t off = 0;
    float* h_ws  = (float*)(ws + off); off += (size_t)NTOK * HID * sizeof(float);   // 45.1 MB
    int*   cnt   = (int*)(ws + off);   off += 256;
    float* probs = (float*)(ws + off); off += (size_t)NTOK * N_EXP * sizeof(float); // 128 KB
    float* top1  = (float*)(ws + off); off += (size_t)NTOK * sizeof(float);         // 16 KB
    int*   lists = (int*)(ws + off);   off += (size_t)N_EXP * NTOK * sizeof(int);   // 128 KB

    moe_init_kernel<<<1, 64, 0, stream>>>(cnt);
    moe_router_kernel<<<NTOK / 8, 256, 0, stream>>>(x, Wg, bg, cnt, probs, top1, lists);
    moe_aux_kernel<<<1, 256, 0, stream>>>(cnt, probs, out + (size_t)NTOK * D_MODEL);
    moe_up_kernel<<<dim3(HID / BN, NTOK / BM, N_EXP), 128, 0, stream>>>(x, Wu, Wv, cnt, lists, h_ws);
    moe_down_kernel<<<dim3(D_MODEL / BN, NTOK / BM, N_EXP), 128, 0, stream>>>(h_ws, Wd, cnt, lists, top1, out);
}
